// MLA_72258529788653
// MI455X (gfx1250) — matrix operand order repaired and checked by reading, not compiled, not run on hardware
//
#include <hip/hip_runtime.h>

// ---------------- types ----------------
typedef float  v2f  __attribute__((ext_vector_type(2)));
typedef float  v4f  __attribute__((ext_vector_type(4)));
typedef float  v8f  __attribute__((ext_vector_type(8)));
typedef __bf16 bf16;
typedef __bf16 v4bf  __attribute__((ext_vector_type(4)));
typedef __bf16 v8bf  __attribute__((ext_vector_type(8)));
typedef __bf16 v16bf __attribute__((ext_vector_type(16)));

// ---------------- problem constants ----------------
#define B_     32
#define H_     128
#define DIM_   7168
#define QLR_   1536
#define KVLR_  512
#define DN_    128
#define DR_    64
#define DV_    128
#define QKD_   192     // DN + DR
#define SD_    576     // KVLR + DR   (absorbed score feature dim)
#define SD4_   144     // SD_/4
#define MAXS_  4096
#define SPOS_  4095
#define NSPLIT 8
#define CHUNK  512
#define TT     32
#define SCALE_ 0.07216878364870323f  // 1/sqrt(192)
#define EPS_   1e-6f

// ---------------- helpers ----------------
static __device__ inline v8f v8f_zero() {
  v8f z;
#pragma unroll
  for (int i = 0; i < 8; ++i) z[i] = 0.f;
  return z;
}

static __device__ inline v16bf cat8(v8bf a, v8bf b) {
  return __builtin_shufflevector(a, b, 0,1,2,3,4,5,6,7,8,9,10,11,12,13,14,15);
}

static __device__ inline v8f wmma_f32x4(v2f a, v2f b, v8f c) {
  // D = A(16x4 f32) * B(4x16 f32) + C
  return __builtin_amdgcn_wmma_f32_16x16x4_f32(false, a, false, b, (short)0, c, false, false);
}

static __device__ inline v8f wmma_bf16(v16bf a, v16bf b, v8f c) {
  // D = A(16x32 bf16) * B(32x16 bf16) + C(f32)
  return __builtin_amdgcn_wmma_f32_16x16x32_bf16(false, a, false, b, (short)0, c, false, false);
}

// CDNA5 async copy: 16B per lane, global -> LDS, tracked by ASYNCcnt.
static __device__ inline void async_b128(void* lds_dst, const float* gsrc) {
  unsigned l = (unsigned)(size_t)lds_dst;   // low 32 bits of a generic LDS ptr = LDS offset
  asm volatile("global_load_async_to_lds_b128 %0, %1, off"
               :: "v"(l), "v"(gsrc) : "memory");
}

// =====================================================================
// Generic f32 WMMA GEMM:  C(32 x N) = A(32 x K) * W(N x K)^T
// block = 64 threads (2 waves; wave w computes rows 16w..16w+15).
// W tiles (16 x 64 f32 = 4KB) stream through a 4-deep LDS pipeline fed
// by global_load_async_to_lds_b128; WMMA B-operands come from LDS.
// grid = (N/16, 1, nbatch).  K must be a multiple of 64 (true here).
// =====================================================================
__global__ __launch_bounds__(64, 1) void gemm_xt_f32(
    const float* __restrict__ A, const float* __restrict__ W,
    float* __restrict__ C, int K,
    int lda, int ldw, int ldc,
    long aBatch, long wBatch, long cBatch)
{
  __shared__ float Wt[4][16][64];   // 16 KB
  const int tidb = threadIdx.x;     // 0..63
  const int lane = tidb & 31;
  const int wave = tidb >> 5;
  const int hl = lane & 15, g = lane >> 4;
  A += (size_t)blockIdx.z * (size_t)aBatch;
  W += (size_t)blockIdx.z * (size_t)wBatch;
  C += (size_t)blockIdx.z * (size_t)cBatch;
  const int m0 = wave * 16;
  const int n0 = blockIdx.x * 16;
  const float* ap = A + (size_t)(m0 + hl) * (size_t)lda + 2 * g;

  // each thread copies 4 x 16B chunks per tile: chunk c = tidb + 64*j,
  // row = c>>4 (16 chunks of 16B per 64-float row), col = (c&15)*4
#pragma unroll
  for (int t = 0; t < 3; ++t) {
    if (t * 64 < K) {
#pragma unroll
      for (int j = 0; j < 4; ++j) {
        int c = tidb + 64 * j;
        int row = c >> 4, col = (c & 15) * 4;
        async_b128(&Wt[t][row][col],
                   W + (size_t)(n0 + row) * (size_t)ldw + t * 64 + col);
      }
    }
  }

  v8f acc = v8f_zero();
  for (int kb = 0; kb < K; kb += 64) {
    const int ib = (kb >> 6) & 3;
    if (kb + 192 < K) {
#pragma unroll
      for (int j = 0; j < 4; ++j) {
        int c = tidb + 64 * j;
        int row = c >> 4, col = (c & 15) * 4;
        async_b128(&Wt[(ib + 3) & 3][row][col],
                   W + (size_t)(n0 + row) * (size_t)ldw + kb + 192 + col);
      }
      asm volatile("s_wait_asynccnt 0xc" ::: "memory");   // tile kb landed
    } else if (kb + 128 < K) {
      asm volatile("s_wait_asynccnt 0x8" ::: "memory");
    } else if (kb + 64 < K) {
      asm volatile("s_wait_asynccnt 0x4" ::: "memory");
    } else {
      asm volatile("s_wait_asynccnt 0x0" ::: "memory");
    }
    __syncthreads();
#pragma unroll
    for (int k0 = 0; k0 < 64; k0 += 4) {
      v2f a = *(const v2f*)(ap + kb + k0);                 // A: m=lane&15, k pair k0+2g
      v2f w = *(const v2f*)(&Wt[ib][hl][k0 + 2 * g]);      // B from LDS
      acc = wmma_f32x4(a, w, acc);
    }
    __syncthreads();   // readers done before this buffer is refilled
  }
#pragma unroll
  for (int v = 0; v < 8; ++v)
    C[(size_t)(m0 + v + 8 * g) * (size_t)ldc + n0 + hl] = acc[v];
}

// =====================================================================
// Generic f32 WMMA GEMM:  C(32 x N) = A(32 x K) * Bm(K x N)
// (B stored row-major K x N; used for q_abs = q_nope @ wkv_b_r[h,:DN])
// =====================================================================
__global__ __launch_bounds__(64, 1) void gemm_ab_f32(
    const float* __restrict__ A, const float* __restrict__ Bm,
    float* __restrict__ C, int K,
    int lda, int ldb, int ldc,
    long aBatch, long bBatch, long cBatch)
{
  const int lane = threadIdx.x & 31;
  const int wave = threadIdx.x >> 5;
  const int hl = lane & 15, g = lane >> 4;
  A  += (size_t)blockIdx.z * (size_t)aBatch;
  Bm += (size_t)blockIdx.z * (size_t)bBatch;
  C  += (size_t)blockIdx.z * (size_t)cBatch;
  const int m0 = wave * 16;
  const int n0 = blockIdx.x * 16;
  const float* ap = A + (size_t)(m0 + hl) * (size_t)lda + 2 * g;
  const float* bp = Bm + n0 + hl;
  v8f acc = v8f_zero();
  for (int k0 = 0; k0 < K; k0 += 4) {
    v2f a = *(const v2f*)(ap + k0);
    v2f b;
    b.x = bp[(size_t)(k0 + 2 * g) * (size_t)ldb];
    b.y = bp[(size_t)(k0 + 2 * g + 1) * (size_t)ldb];
    acc = wmma_f32x4(a, b, acc);
  }
#pragma unroll
  for (int v = 0; v < 8; ++v)
    C[(size_t)(m0 + v + 8 * g) * (size_t)ldc + n0 + hl] = acc[v];
}

// =====================================================================
// In-place RMSNorm of 32 rows of `ncols`
// =====================================================================
__global__ void rms_rows_k(float* __restrict__ y, const float* __restrict__ w, int ncols)
{
  __shared__ float red[256];
  const int b = blockIdx.x, tid = threadIdx.x;
  float* row = y + (size_t)b * ncols;
  float ss = 0.f;
  for (int i = tid; i < ncols; i += 256) { float v = row[i]; ss += v * v; }
  red[tid] = ss; __syncthreads();
  for (int s = 128; s > 0; s >>= 1) { if (tid < s) red[tid] += red[tid + s]; __syncthreads(); }
  const float rs = rsqrtf(red[0] / (float)ncols + EPS_);
  for (int i = tid; i < ncols; i += 256) row[i] = row[i] * rs * w[i];
}

// =====================================================================
// kv_full(32 x 576) -> kv_new = rmsnorm(kv_full[:,:512]) ; pe_new = rope(kv_full[:,512:])
// =====================================================================
__global__ void kv_prep_k(const float* __restrict__ kvf, const float* __restrict__ kvw,
                          const float* __restrict__ fc, const float* __restrict__ fs,
                          float* __restrict__ kvn, float* __restrict__ pen)
{
  __shared__ float red[256];
  const int b = blockIdx.x, tid = threadIdx.x;
  const float* row = kvf + (size_t)b * (KVLR_ + DR_);
  float ss = 0.f;
  for (int i = tid; i < KVLR_; i += 256) { float v = row[i]; ss += v * v; }
  red[tid] = ss; __syncthreads();
  for (int s = 128; s > 0; s >>= 1) { if (tid < s) red[tid] += red[tid + s]; __syncthreads(); }
  const float rs = rsqrtf(red[0] / (float)KVLR_ + EPS_);
  for (int i = tid; i < KVLR_; i += 256) kvn[(size_t)b * KVLR_ + i] = row[i] * rs * kvw[i];
  if (tid < DR_ / 2) {
    float x1 = row[KVLR_ + 2 * tid], x2 = row[KVLR_ + 2 * tid + 1];
    float c = fc[tid], s = fs[tid];
    pen[(size_t)b * DR_ + 2 * tid]     = x1 * c - x2 * s;
    pen[(size_t)b * DR_ + 2 * tid + 1] = x1 * s + x2 * c;
  }
}

// =====================================================================
// RoPE on q_pe part of q buffer, in place.  grid*block == B*H*DR/2
// =====================================================================
__global__ void rope_q_k(float* __restrict__ q, const float* __restrict__ fc,
                         const float* __restrict__ fs)
{
  int idx = blockIdx.x * blockDim.x + threadIdx.x;   // < 32*128*32
  int b = idx >> 12;
  int r = idx & 4095;
  int h = r >> 5;
  int i = r & 31;
  float* p = q + (size_t)b * (H_ * QKD_) + (size_t)h * QKD_ + DN_ + 2 * i;
  float x1 = p[0], x2 = p[1];
  float c = fc[i], s = fs[i];
  p[0] = x1 * c - x2 * s;
  p[1] = x1 * s + x2 * c;
}

// =====================================================================
// Flash-style MLA attention over one 512-position chunk of one batch.
// grid (NSPLIT, B), block 512 (16 wave32).
// Waves 0..7: score+softmax for head-tile w (16 heads, 576-deep QK via
//   bf16 WMMA), write P (bf16) and per-row alpha to LDS.
// All 16 waves: O update; wave w owns head-tile w>>1, column half (w&1)*256.
// =====================================================================
__global__ __launch_bounds__(512, 1) void attn_k(
    const float* __restrict__ qabs, const float* __restrict__ q,
    const float* __restrict__ kvc, const float* __restrict__ pec,
    const float* __restrict__ kvn, const float* __restrict__ pen,
    float* __restrict__ Opart, float* __restrict__ Mst, float* __restrict__ Lst)
{
  extern __shared__ char smem[];
  bf16*  Qs     = (bf16*)smem;                 // [128][576]
  bf16*  KVs    = Qs + H_ * SD_;               // [32][576]  (kv | pe)
  bf16*  KVsT   = KVs + TT * SD_;              // [512][32]  transposed V
  bf16*  Ps     = KVsT + KVLR_ * TT;           // [128][32]  P tiles
  float* alphaS = (float*)(Ps + H_ * TT);      // [128]

  const int b = blockIdx.y, chunk = blockIdx.x;
  const int tid = threadIdx.x;
  const int lane = tid & 31, wave = tid >> 5;
  const int hl = lane & 15, g = lane >> 4;

  // ---- stage Q (q_abs | roped q_pe) as bf16, float4 at a time ----
  for (int idx = tid; idx < H_ * SD4_; idx += 512) {
    int h = idx / SD4_, cc = (idx - h * SD4_) * 4;
    v4f f;
    if (cc < KVLR_)
      f = *(const v4f*)(qabs + ((size_t)b * H_ + h) * KVLR_ + cc);
    else
      f = *(const v4f*)(q + (size_t)b * (H_ * QKD_) + (size_t)h * QKD_ + DN_ + (cc - KVLR_));
    v4bf hb;
#pragma unroll
    for (int e = 0; e < 4; ++e) hb[e] = (bf16)f[e];
    *(v4bf*)(Qs + h * SD_ + cc) = hb;
  }
  __syncthreads();

  const int ht = wave >> 1;   // O head-tile
  const int ch = wave & 1;    // O column half (256 cols)
  v8f accO[16];
#pragma unroll
  for (int nt = 0; nt < 16; ++nt) accO[nt] = v8f_zero();

  float mrow[8], lrow[8];
#pragma unroll
  for (int v = 0; v < 8; ++v) { mrow[v] = -3.0e38f; lrow[v] = 0.f; }

  for (int tb = 0; tb < CHUNK; tb += TT) {
    const int t0 = chunk * CHUNK + tb;
    // ---- stage K/V tile (576-wide rows; t==SPOS from fresh kv/pe) ----
    for (int idx = tid; idx < TT * SD4_; idx += 512) {
      int tl = idx / SD4_, cc = (idx - tl * SD4_) * 4;
      int tg = t0 + tl;
      v4f f;
      if (cc < KVLR_)
        f = (tg == SPOS_) ? *(const v4f*)(kvn + (size_t)b * KVLR_ + cc)
                          : *(const v4f*)(kvc + ((size_t)b * MAXS_ + tg) * KVLR_ + cc);
      else {
        int rr = cc - KVLR_;
        f = (tg == SPOS_) ? *(const v4f*)(pen + (size_t)b * DR_ + rr)
                          : *(const v4f*)(pec + ((size_t)b * MAXS_ + tg) * DR_ + rr);
      }
      v4bf hb;
#pragma unroll
      for (int e = 0; e < 4; ++e) hb[e] = (bf16)f[e];
      *(v4bf*)(KVs + tl * SD_ + cc) = hb;
      if (cc < KVLR_) {
#pragma unroll
        for (int e = 0; e < 4; ++e) KVsT[(cc + e) * TT + tl] = hb[e];
      }
    }
    __syncthreads();

    if (wave < 8) {
      // ---- scores: two 16x16 tiles over 576-deep K ----
      const bf16* qrow = Qs + (size_t)(wave * 16 + hl) * SD_;
      v8f S[2];
#pragma unroll
      for (int j = 0; j < 2; ++j) {
        v8f s = v8f_zero();
        const bf16* krow = KVs + (size_t)(j * 16 + hl) * SD_;
#pragma unroll
        for (int k0 = 0; k0 < SD_; k0 += 32) {
          v16bf a  = cat8(*(const v8bf*)(qrow + k0 + g * 8),
                          *(const v8bf*)(qrow + k0 + 16 + g * 8));
          v16bf kb = cat8(*(const v8bf*)(krow + k0 + g * 8),
                          *(const v8bf*)(krow + k0 + 16 + g * 8));
          s = wmma_bf16(a, kb, s);
        }
        S[j] = s;
      }
      // ---- online softmax per row; emit P (bf16) and alpha ----
      bf16* pw = Ps + wave * 16 * TT;
#pragma unroll
      for (int v = 0; v < 8; ++v) {
        float s0 = S[0][v] * SCALE_;
        float s1 = S[1][v] * SCALE_;
        float mx = fmaxf(s0, s1);
#pragma unroll
        for (int off = 8; off >= 1; off >>= 1)
          mx = fmaxf(mx, __shfl_xor(mx, off, 32));
        float mnew = fmaxf(mrow[v], mx);
        float al = __expf(mrow[v] - mnew);
        float p0 = __expf(s0 - mnew);
        float p1 = __expf(s1 - mnew);
        float rsum = p0 + p1;
#pragma unroll
        for (int off = 8; off >= 1; off >>= 1)
          rsum += __shfl_xor(rsum, off, 32);
        lrow[v] = lrow[v] * al + rsum;
        mrow[v] = mnew;
        const int rowr = v + 8 * g;
        pw[rowr * TT + hl]      = (bf16)p0;
        pw[rowr * TT + 16 + hl] = (bf16)p1;
        if (hl == 0) alphaS[wave * 16 + rowr] = al;
      }
    }
    __syncthreads();

    // ---- O update: accO = accO*alpha + P(16x32) x V(32x16-col-tiles) ----
    {
      const bf16* pw = Ps + ht * 16 * TT;
      v16bf pa = cat8(*(const v8bf*)(pw + hl * TT + g * 8),
                      *(const v8bf*)(pw + hl * TT + 16 + g * 8));
      float al[8];
#pragma unroll
      for (int v = 0; v < 8; ++v) al[v] = alphaS[ht * 16 + v + 8 * g];
#pragma unroll
      for (int nt = 0; nt < 16; ++nt) {
#pragma unroll
        for (int v = 0; v < 8; ++v) accO[nt][v] *= al[v];
        const bf16* vcol = KVsT + (size_t)(ch * 256 + nt * 16 + hl) * TT;
        v16bf vb = cat8(*(const v8bf*)(vcol + g * 8),
                        *(const v8bf*)(vcol + 16 + g * 8));
        accO[nt] = wmma_bf16(pa, vb, accO[nt]);
      }
    }
    __syncthreads();
  }

  // ---- epilogue: partial O + (m,l) stats ----
  {
    size_t base = (((size_t)b * NSPLIT + chunk) * H_ + ht * 16) * KVLR_;
#pragma unroll
    for (int nt = 0; nt < 16; ++nt)
#pragma unroll
      for (int v = 0; v < 8; ++v)
        Opart[base + (size_t)(v + 8 * g) * KVLR_ + ch * 256 + nt * 16 + hl] = accO[nt][v];
  }
  if (wave < 8 && hl == 0) {
    size_t sb = ((size_t)b * NSPLIT + chunk) * H_ + wave * 16;
#pragma unroll
    for (int v = 0; v < 8; ++v) {
      Mst[sb + v + 8 * g] = mrow[v];
      Lst[sb + v + 8 * g] = lrow[v];
    }
  }
}

// =====================================================================
// Combine NSPLIT partial attention results.  grid (H, B), block 128.
// =====================================================================
__global__ void combine_k(const float* __restrict__ Opart,
                          const float* __restrict__ Mst, const float* __restrict__ Lst,
                          float* __restrict__ oh)
{
  const int h = blockIdx.x, b = blockIdx.y, tid = threadIdx.x;
  float m[NSPLIT], M = -3.0e38f;
#pragma unroll
  for (int k = 0; k < NSPLIT; ++k) {
    m[k] = Mst[((size_t)b * NSPLIT + k) * H_ + h];
    M = fmaxf(M, m[k]);
  }
  float wgt[NSPLIT], L = 0.f;
#pragma unroll
  for (int k = 0; k < NSPLIT; ++k) {
    wgt[k] = __expf(m[k] - M);
    L += Lst[((size_t)b * NSPLIT + k) * H_ + h] * wgt[k];
  }
  const float invL = 1.f / L;
  for (int c = tid; c < KVLR_; c += blockDim.x) {
    float acc = 0.f;
#pragma unroll
    for (int k = 0; k < NSPLIT; ++k)
      acc += wgt[k] * Opart[(((size_t)b * NSPLIT + k) * H_ + h) * KVLR_ + c];
    oh[((size_t)b * H_ + h) * KVLR_ + c] = acc * invL;
  }
}

// =====================================================================
extern "C" void kernel_launch(void* const* d_in, const int* in_sizes, int n_in,
                              void* d_out, int out_size, void* d_ws, size_t ws_size,
                              hipStream_t stream)
{
  (void)in_sizes; (void)n_in; (void)out_size; (void)ws_size;
  const float* x    = (const float*)d_in[0];
  const float* fc   = (const float*)d_in[1];
  const float* fs   = (const float*)d_in[2];
  const float* kvc  = (const float*)d_in[3];
  const float* pec  = (const float*)d_in[4];
  const float* wqa  = (const float*)d_in[5];
  const float* qnw  = (const float*)d_in[6];
  const float* wqb  = (const float*)d_in[7];
  const float* wkva = (const float*)d_in[8];
  const float* kvnw = (const float*)d_in[9];
  const float* wkvb = (const float*)d_in[10];
  const float* wo   = (const float*)d_in[11];
  // d_in[12] = start_pos (fixed at 4095 by the harness shapes)

  float* ws = (float*)d_ws;
  size_t off = 0;
  float* q_lat   = ws + off; off += (size_t)B_ * QLR_;
  float* kv_full = ws + off; off += (size_t)B_ * (KVLR_ + DR_);
  float* kv_new  = ws + off; off += (size_t)B_ * KVLR_;
  float* pe_new  = ws + off; off += (size_t)B_ * DR_;
  float* qbuf    = ws + off; off += (size_t)B_ * H_ * QKD_;
  float* q_abs   = ws + off; off += (size_t)B_ * H_ * KVLR_;
  float* o_heads = ws + off; off += (size_t)B_ * H_ * KVLR_;
  float* o2      = ws + off; off += (size_t)B_ * H_ * DV_;
  float* Mst     = ws + off; off += (size_t)B_ * NSPLIT * H_;
  float* Lst     = ws + off; off += (size_t)B_ * NSPLIT * H_;
  float* Opart   = ws + off; off += (size_t)B_ * NSPLIT * H_ * KVLR_;

  // 1. q_lat_raw = x @ wq_a^T   (32x1536, K=7168)
  gemm_xt_f32<<<dim3(QLR_ / 16, 1, 1), 64, 0, stream>>>(
      x, wqa, q_lat, DIM_, DIM_, DIM_, QLR_, 0, 0, 0);
  // 2. kv_full = x @ wkv_a^T    (32x576, K=7168)
  gemm_xt_f32<<<dim3((KVLR_ + DR_) / 16, 1, 1), 64, 0, stream>>>(
      x, wkva, kv_full, DIM_, DIM_, DIM_, KVLR_ + DR_, 0, 0, 0);
  // 3. q_lat = rmsnorm(q_lat_raw)
  rms_rows_k<<<B_, 256, 0, stream>>>(q_lat, qnw, QLR_);
  // 4. kv_new = rmsnorm(kv), pe_new = rope(k_pe)
  kv_prep_k<<<B_, 256, 0, stream>>>(kv_full, kvnw, fc, fs, kv_new, pe_new);
  // 5. q = q_lat @ wq_b^T       (32x24576, K=1536)
  gemm_xt_f32<<<dim3(H_ * QKD_ / 16, 1, 1), 64, 0, stream>>>(
      q_lat, wqb, qbuf, QLR_, QLR_, QLR_, H_ * QKD_, 0, 0, 0);
  // 6. rope(q_pe) in place
  rope_q_k<<<(B_ * H_ * (DR_ / 2)) / 256, 256, 0, stream>>>(qbuf, fc, fs);
  // 7. q_abs[b,h] = q_nope[b,h] @ wkv_b_r[h,:DN]   (per-head, 32x512, K=128)
  gemm_ab_f32<<<dim3(KVLR_ / 16, 1, H_), 64, 0, stream>>>(
      qbuf, wkvb, q_abs, DN_,
      H_ * QKD_, KVLR_, H_ * KVLR_,
      (long)QKD_, (long)((DN_ + DV_) * KVLR_), (long)KVLR_);
  // 8. flash attention over t-chunks
  const size_t lds = (size_t)(H_ * SD_ + TT * SD_ + KVLR_ * TT + H_ * TT) * sizeof(bf16)
                   + (size_t)H_ * sizeof(float);   // 225792 B
  attn_k<<<dim3(NSPLIT, B_), 512, lds, stream>>>(
      q_abs, qbuf, kvc, pec, kv_new, pe_new, Opart, Mst, Lst);
  // 9. combine partials
  combine_k<<<dim3(H_, B_), 128, 0, stream>>>(Opart, Mst, Lst, o_heads);
  // 10. o2[b,h] = o_heads[b,h] @ wkv_b_r[h,DN:]^T  (per-head, 32x128, K=512)
  gemm_xt_f32<<<dim3(DV_ / 16, 1, H_), 64, 0, stream>>>(
      o_heads, wkvb + (size_t)DN_ * KVLR_, o2, KVLR_,
      H_ * KVLR_, KVLR_, H_ * DV_,
      (long)KVLR_, (long)((DN_ + DV_) * KVLR_), (long)DV_);
  // 11. out = o2 @ wo^T         (32x7168, K=16384)
  gemm_xt_f32<<<dim3(DIM_ / 16, 1, 1), 64, 0, stream>>>(
      o2, wo, (float*)d_out, H_ * DV_, H_ * DV_, H_ * DV_, DIM_, 0, 0, 0);
}
